// SPHNet_25451976196779
// MI455X (gfx1250) — matrix operand order repaired and checked by reading, not compiled, not run on hardware
//
#include <hip/hip_runtime.h>

typedef __attribute__((ext_vector_type(16))) _Float16 v16h;
typedef __attribute__((ext_vector_type(8)))  float    v8f;

#define DIM 512   // 128 scalar + 128*3 vector

// ---------------------------------------------------------------------------
// WMMA helpers (CDNA5 wave32, 16x16x32 f16 -> f32 accumulate)
// ---------------------------------------------------------------------------
__device__ __forceinline__ v8f wmma16(v16h a, v16h b, v8f c) {
  // (neg_a, A, neg_b, B, c_mod, C, reuse_a, reuse_b)
  return __builtin_amdgcn_wmma_f32_16x16x32_f16(false, a, false, b, (short)0, c,
                                                false, false);
}

// A fragment (16x32 f16, row-major in LDS with row stride `ld` halves).
// Contiguous k order used here (matches pack_w_kernel): lanes 0-15 hold row M=lane, K in {0..15};
//             lanes 16-31 hold row M=lane-16, K in {16..31}.
__device__ __forceinline__ v16h lda_frag(const _Float16* base, int ld, int lane) {
  const int row = lane & 15;
  const int kb  = (lane >> 4) * 16;
  const _Float16* p = base + row * ld + kb;
  v16h a;
#pragma unroll
  for (int i = 0; i < 8; ++i) { a[i] = p[i]; a[i + 8] = p[i + 8]; }
  return a;
}

// B fragment: pre-packed per-lane (32 lanes x 16 halves contiguous per tile).
__device__ __forceinline__ v16h ldb_frag(const _Float16* __restrict__ packed,
                                         int tile, int lane) {
  return ((const v16h*)packed)[(size_t)tile * 32 + lane];
}

__device__ __forceinline__ float sspf(float z) {  // softplus(z) - ln 2
  float sp = (z > 15.f) ? z : log1pf(__expf(z));
  return sp - 0.69314718055994531f;
}

// ---------------------------------------------------------------------------
// Pack W (row-major K x N, fp32) into WMMA B-fragment layout, scaled, f16.
// tileIdx = nt*numKt + kt ; per lane: col = lane&15, kbase = (lane>>4)*16,
// element i <-> K = kt*32 + kbase + i.
// ---------------------------------------------------------------------------
__global__ void pack_w_kernel(const float* __restrict__ src,
                              _Float16* __restrict__ dst,
                              int K, int Nout, float scale) {
  int t = blockIdx.x * blockDim.x + threadIdx.x;
  int numKt = K >> 5;
  int total = numKt * (Nout >> 4) * 32;
  if (t >= total) return;
  int lane = t & 31, tile = t >> 5;
  int kt = tile % numKt, nt = tile / numKt;
  int col = lane & 15, kb = (lane >> 4) * 16;
  const float* s = src + (size_t)(kt * 32 + kb) * Nout + nt * 16 + col;
  _Float16* d = dst + (size_t)tile * 512 + lane * 16;
#pragma unroll
  for (int i = 0; i < 16; ++i) d[i] = (_Float16)(s[(size_t)i * Nout] * scale);
}

// ---------------------------------------------------------------------------
// Node kernel: pre_x = irrep_linear(x, Wp0,bp0,Wp1)  (planar [s|vx|vy|vz])
//              self_x = irrep_linear(norm_gate(x), Wn0,bn0,Wn1) (planar)
//              out_acc = self_x (reference interleaved layout)
// One block = 16 nodes, 8 waves; each wave owns a 16-wide output column tile.
// FULL=true blocks compile with no bounds checks (N multiple of 16 hot path).
// ---------------------------------------------------------------------------
template <bool FULL>
__global__ __launch_bounds__(256) void node_kernel(
    const float* __restrict__ x,
    float* __restrict__ pre_x, float* __restrict__ self_x,
    float* __restrict__ out_acc,
    const _Float16* __restrict__ pWp0, const _Float16* __restrict__ pWp1,
    const _Float16* __restrict__ pngW1, const _Float16* __restrict__ pngW2,
    const _Float16* __restrict__ pWn0, const _Float16* __restrict__ pWn1,
    const float* __restrict__ bp0, const float* __restrict__ ngb1,
    const float* __restrict__ ngb2, const float* __restrict__ bn0,
    int tile0, int Nn) {
  __shared__ _Float16 sF0[16][256];       // [s | norm_v]
  __shared__ _Float16 sV[3][16][128];     // planar vector parts of x
  __shared__ _Float16 sH[16][256];        // silu hidden
  __shared__ _Float16 sG[16][512];        // gated output, planar

  const int tid = threadIdx.x;
  const int lane = tid & 31;
  const int wv = tid >> 5;
  const int n0 = (tile0 + blockIdx.x) * 16;

  for (int it = tid; it < 16 * 128; it += 256) {
    int row = it >> 7, u = it & 127;
    int n = n0 + row;
    if (!FULL && n >= Nn) n = Nn - 1;
    const float* xr = x + (size_t)n * DIM;
    sF0[row][u] = (_Float16)xr[u];
    float v0 = xr[128 + u * 3 + 0];
    float v1 = xr[128 + u * 3 + 1];
    float v2 = xr[128 + u * 3 + 2];
    sV[0][row][u] = (_Float16)v0;
    sV[1][row][u] = (_Float16)v1;
    sV[2][row][u] = (_Float16)v2;
    sF0[row][128 + u] = (_Float16)sqrtf(v0 * v0 + v1 * v1 + v2 * v2 + 1e-12f);
  }
  __syncthreads();

  const int col = lane & 15, rb = (lane >> 4) * 8;

  // ---- pre_x = irrep_linear(x, Wp0, bp0, Wp1) ----
  {
    v8f c = {};
#pragma unroll
    for (int kt = 0; kt < 4; ++kt)
      c = wmma16(lda_frag(&sF0[0][0] + kt * 32, 256, lane),
                 ldb_frag(pWp0, wv * 4 + kt, lane), c);
    float bias = bp0[wv * 16 + col];
#pragma unroll
    for (int r = 0; r < 8; ++r) {
      int n = n0 + rb + r;
      if (FULL || n < Nn) pre_x[(size_t)n * DIM + wv * 16 + col] = c[r] + bias;
    }
#pragma unroll
    for (int m = 0; m < 3; ++m) {
      v8f cv = {};
#pragma unroll
      for (int kt = 0; kt < 4; ++kt)
        cv = wmma16(lda_frag(&sV[m][0][0] + kt * 32, 128, lane),
                    ldb_frag(pWp1, wv * 4 + kt, lane), cv);
#pragma unroll
      for (int r = 0; r < 8; ++r) {
        int n = n0 + rb + r;
        if (FULL || n < Nn)
          pre_x[(size_t)n * DIM + 128 + m * 128 + wv * 16 + col] = cv[r];
      }
    }
  }

  // ---- norm_gate hidden: h = silu(f0 @ ngW1 + b1) ----
#pragma unroll
  for (int j = 0; j < 2; ++j) {
    int nt = wv + j * 8;
    v8f c = {};
#pragma unroll
    for (int kt = 0; kt < 8; ++kt)
      c = wmma16(lda_frag(&sF0[0][0] + kt * 32, 256, lane),
                 ldb_frag(pngW1, nt * 8 + kt, lane), c);
#pragma unroll
    for (int r = 0; r < 8; ++r) {
      float z = c[r] + ngb1[nt * 16 + col];
      sH[rb + r][nt * 16 + col] = (_Float16)(z / (1.f + __expf(-z)));
    }
  }
  __syncthreads();

  // ---- g = h @ ngW2 + b2 ; gate vectors ----
#pragma unroll
  for (int j = 0; j < 2; ++j) {
    int nt = wv + j * 8;
    v8f c = {};
#pragma unroll
    for (int kt = 0; kt < 8; ++kt)
      c = wmma16(lda_frag(&sH[0][0] + kt * 32, 256, lane),
                 ldb_frag(pngW2, nt * 8 + kt, lane), c);
#pragma unroll
    for (int r = 0; r < 8; ++r) {
      int cc = nt * 16 + col, row = rb + r;
      float g = c[r] + ngb2[cc];
      if (cc < 128) {
        sG[row][cc] = (_Float16)g;
      } else {
        int u = cc - 128;
        sG[row][128 + 0 * 128 + u] = (_Float16)((float)sV[0][row][u] * g);
        sG[row][128 + 1 * 128 + u] = (_Float16)((float)sV[1][row][u] * g);
        sG[row][128 + 2 * 128 + u] = (_Float16)((float)sV[2][row][u] * g);
      }
    }
  }
  __syncthreads();

  // ---- self_x = irrep_linear(gated, Wn0, bn0, Wn1); out_acc = self_x ----
  {
    v8f c = {};
#pragma unroll
    for (int kt = 0; kt < 4; ++kt)
      c = wmma16(lda_frag(&sG[0][0] + kt * 32, 512, lane),
                 ldb_frag(pWn0, wv * 4 + kt, lane), c);
    float bias = bn0[wv * 16 + col];
#pragma unroll
    for (int r = 0; r < 8; ++r) {
      int n = n0 + rb + r;
      if (FULL || n < Nn) {
        float val = c[r] + bias;
        self_x[(size_t)n * DIM + wv * 16 + col] = val;
        out_acc[(size_t)n * DIM + wv * 16 + col] = val;
      }
    }
#pragma unroll
    for (int m = 0; m < 3; ++m) {
      v8f cv = {};
#pragma unroll
      for (int kt = 0; kt < 4; ++kt)
        cv = wmma16(lda_frag(&sG[0][0] + 128 + m * 128 + kt * 32, 512, lane),
                    ldb_frag(pWn1, wv * 4 + kt, lane), cv);
#pragma unroll
      for (int r = 0; r < 8; ++r) {
        int n = n0 + rb + r;
        if (FULL || n < Nn) {
          int u = wv * 16 + col;
          self_x[(size_t)n * DIM + 128 + m * 128 + u] = cv[r];
          out_acc[(size_t)n * DIM + 128 + u * 3 + m] = cv[r];
        }
      }
    }
  }
}

// ---------------------------------------------------------------------------
// Edge kernel: one block = 16 edges, 8 waves.
//   s0 = [pre_s(dst), pre_s(src), <v_dst,v_src>/sqrt3]
//   w  = mlp2(edge_attr, fcnW1, fcnW2) * mlp2(s0, l0W1, l0W2)   (WMMA)
//   gated tensor product -> atomic fadd scatter into out_acc[dst]
// ---------------------------------------------------------------------------
template <bool FULL>
__global__ __launch_bounds__(256) void edge_kernel(
    const int* __restrict__ edge_index, const float* __restrict__ edge_attr,
    const float* __restrict__ edge_sh,
    const float* __restrict__ pre_x, const float* __restrict__ self_x,
    float* __restrict__ out_acc,
    const _Float16* __restrict__ pFcn1, const _Float16* __restrict__ pFcn2,
    const _Float16* __restrict__ pL01, const _Float16* __restrict__ pL02,
    int tile0, int E) {
  __shared__ int sDst[16], sSrc[16];
  __shared__ float sSh[16][4];
  __shared__ _Float16 sAttr[16][32];
  __shared__ _Float16 sS0[16][384];
  __shared__ _Float16 sXs[4][16][128];   // self_x[src]: [s, vx, vy, vz]
  __shared__ _Float16 sHf[16][32], sHl[16][32];
  __shared__ float sW[16][512];

  const int tid = threadIdx.x, lane = tid & 31, wv = tid >> 5;
  const int e0 = (tile0 + blockIdx.x) * 16;

  if (tid < 16) {
    int e = e0 + tid;
    if (!FULL && e >= E) e = E - 1;
    sDst[tid] = edge_index[e];
    sSrc[tid] = edge_index[(size_t)E + e];
    __builtin_prefetch(&self_x[(size_t)edge_index[(size_t)E + e] * DIM], 0, 1);
  }
  if (tid >= 32 && tid < 96) {
    int t = tid - 32;
    int e = e0 + (t >> 2);
    if (!FULL && e >= E) e = E - 1;
    sSh[t >> 2][t & 3] = edge_sh[(size_t)e * 4 + (t & 3)];
  }
  if (tid >= 96) {
    for (int it = tid - 96; it < 512; it += 160) {
      int row = it >> 5, cidx = it & 31;
      int e = e0 + row;
      if (!FULL && e >= E) e = E - 1;
      sAttr[row][cidx] = (_Float16)edge_attr[(size_t)e * 32 + cidx];
    }
  }
  __syncthreads();

  for (int it = tid; it < 16 * 128; it += 256) {
    int row = it >> 7, u = it & 127;
    size_t d = (size_t)sDst[row] * DIM;
    size_t s = (size_t)sSrc[row] * DIM;
    sS0[row][u]       = (_Float16)pre_x[d + u];
    sS0[row][128 + u] = (_Float16)pre_x[s + u];
    float ip = pre_x[d + 128 + u] * pre_x[s + 128 + u]
             + pre_x[d + 256 + u] * pre_x[s + 256 + u]
             + pre_x[d + 384 + u] * pre_x[s + 384 + u];
    sS0[row][256 + u] = (_Float16)(ip * 0.57735026918962576f);
    sXs[0][row][u] = (_Float16)self_x[s + u];
    sXs[1][row][u] = (_Float16)self_x[s + 128 + u];
    sXs[2][row][u] = (_Float16)self_x[s + 256 + u];
    sXs[3][row][u] = (_Float16)self_x[s + 384 + u];
  }
  __syncthreads();

  const int col = lane & 15, rb = (lane >> 4) * 8;

  // ---- hidden layers of the two edge MLPs ----
  if (wv < 2) {                         // ssp(s0 @ l0W1/sqrt(384)), N=32
    v8f c = {};
#pragma unroll
    for (int kt = 0; kt < 12; ++kt)
      c = wmma16(lda_frag(&sS0[0][0] + kt * 32, 384, lane),
                 ldb_frag(pL01, wv * 12 + kt, lane), c);
#pragma unroll
    for (int r = 0; r < 8; ++r)
      sHl[rb + r][wv * 16 + col] = (_Float16)sspf(c[r]);
  } else if (wv < 4) {                  // ssp(edge_attr @ fcnW1/sqrt(32)), N=32
    int nt = wv - 2;
    v8f c = {};
    c = wmma16(lda_frag(&sAttr[0][0], 32, lane), ldb_frag(pFcn1, nt, lane), c);
#pragma unroll
    for (int r = 0; r < 8; ++r)
      sHf[rb + r][nt * 16 + col] = (_Float16)sspf(c[r]);
  }
  __syncthreads();

  // ---- second layers (K=32, N=512), elementwise product -> w ----
#pragma unroll
  for (int j = 0; j < 4; ++j) {
    int nt = wv * 4 + j;
    v8f cf = {}, cl = {};
    cf = wmma16(lda_frag(&sHf[0][0], 32, lane), ldb_frag(pFcn2, nt, lane), cf);
    cl = wmma16(lda_frag(&sHl[0][0], 32, lane), ldb_frag(pL02, nt, lane), cl);
#pragma unroll
    for (int r = 0; r < 8; ++r)
      sW[rb + r][nt * 16 + col] = cf[r] * cl[r];
  }
  __syncthreads();

  // ---- gated tensor product + scatter-add (hardware f32 atomics) ----
  for (int it = tid; it < 16 * 128; it += 256) {
    int row = it >> 7, u = it & 127;
    if (!FULL && (e0 + row) >= E) continue;
    float w1 = sW[row][u],       w2 = sW[row][128 + u];
    float w3 = sW[row][256 + u], w4 = sW[row][384 + u];
    float xs0 = (float)sXs[0][row][u];
    float x10 = (float)sXs[1][row][u];
    float x11 = (float)sXs[2][row][u];
    float x12 = (float)sXs[3][row][u];
    float sh0 = sSh[row][0], shx = sSh[row][1];
    float shy = sSh[row][2], shz = sSh[row][3];
    float dot = (x10 * shx + x11 * shy + x12 * shz) * 0.57735026918962576f;
    float o0 = 0.5f * (w1 * xs0 * sh0 + w4 * dot);
    const float c1 = 0.86602540378443865f;      // sqrt(3)/2
    float a = w2 * xs0, b = w3 * sh0;
    float o10 = c1 * (a * shx + b * x10);
    float o11 = c1 * (a * shy + b * x11);
    float o12 = c1 * (a * shz + b * x12);
    float* ob = out_acc + (size_t)sDst[row] * DIM;
    unsafeAtomicAdd(ob + u, o0);
    unsafeAtomicAdd(ob + 128 + u * 3 + 0, o10);
    unsafeAtomicAdd(ob + 128 + u * 3 + 1, o11);
    unsafeAtomicAdd(ob + 128 + u * 3 + 2, o12);
  }
}

// ---------------------------------------------------------------------------
// Output kernel: d_out = irrep_linear(out_acc, Wo0, bo0, Wo1)
// ---------------------------------------------------------------------------
template <bool FULL>
__global__ __launch_bounds__(256) void out_kernel(
    const float* __restrict__ acc, float* __restrict__ out,
    const _Float16* __restrict__ pWo0, const _Float16* __restrict__ pWo1,
    const float* __restrict__ bo0, int tile0, int Nn) {
  __shared__ _Float16 sS[16][128];
  __shared__ _Float16 sV[3][16][128];
  const int tid = threadIdx.x, lane = tid & 31, wv = tid >> 5;
  const int n0 = (tile0 + blockIdx.x) * 16;

  for (int it = tid; it < 16 * 128; it += 256) {
    int row = it >> 7, u = it & 127;
    int n = n0 + row;
    if (!FULL && n >= Nn) n = Nn - 1;
    const float* ar = acc + (size_t)n * DIM;
    sS[row][u]    = (_Float16)ar[u];
    sV[0][row][u] = (_Float16)ar[128 + u * 3 + 0];
    sV[1][row][u] = (_Float16)ar[128 + u * 3 + 1];
    sV[2][row][u] = (_Float16)ar[128 + u * 3 + 2];
  }
  __syncthreads();

  const int col = lane & 15, rb = (lane >> 4) * 8;
  v8f c = {};
#pragma unroll
  for (int kt = 0; kt < 4; ++kt)
    c = wmma16(lda_frag(&sS[0][0] + kt * 32, 128, lane),
               ldb_frag(pWo0, wv * 4 + kt, lane), c);
  float bias = bo0[wv * 16 + col];
#pragma unroll
  for (int r = 0; r < 8; ++r) {
    int n = n0 + rb + r;
    if (FULL || n < Nn) out[(size_t)n * DIM + wv * 16 + col] = c[r] + bias;
  }
#pragma unroll
  for (int m = 0; m < 3; ++m) {
    v8f cv = {};
#pragma unroll
    for (int kt = 0; kt < 4; ++kt)
      cv = wmma16(lda_frag(&sV[m][0][0] + kt * 32, 128, lane),
                  ldb_frag(pWo1, wv * 4 + kt, lane), cv);
#pragma unroll
    for (int r = 0; r < 8; ++r) {
      int n = n0 + rb + r;
      if (FULL || n < Nn)
        out[(size_t)n * DIM + 128 + (wv * 16 + col) * 3 + m] = cv[r];
    }
  }
}

// ---------------------------------------------------------------------------
extern "C" void kernel_launch(void* const* d_in, const int* in_sizes, int n_in,
                              void* d_out, int out_size, void* d_ws,
                              size_t ws_size, hipStream_t stream) {
  const float* x     = (const float*)d_in[0];
  const int*   eidx  = (const int*)d_in[1];
  const float* eattr = (const float*)d_in[2];
  const float* esh   = (const float*)d_in[3];
  const float* Wp0   = (const float*)d_in[4];
  const float* bp0   = (const float*)d_in[5];
  const float* Wp1   = (const float*)d_in[6];
  const float* ngW1  = (const float*)d_in[7];
  const float* ngb1  = (const float*)d_in[8];
  const float* ngW2  = (const float*)d_in[9];
  const float* ngb2  = (const float*)d_in[10];
  const float* Wn0   = (const float*)d_in[11];
  const float* bn0   = (const float*)d_in[12];
  const float* Wn1   = (const float*)d_in[13];
  const float* fcnW1 = (const float*)d_in[14];
  const float* fcnW2 = (const float*)d_in[15];
  const float* l0W1  = (const float*)d_in[16];
  const float* l0W2  = (const float*)d_in[17];
  const float* Wo0   = (const float*)d_in[18];
  const float* bo0   = (const float*)d_in[19];
  const float* Wo1   = (const float*)d_in[20];

  const int Nn = in_sizes[0] / DIM;
  const int E  = in_sizes[1] / 2;

  char* ws = (char*)d_ws;
  auto carve = [&](size_t bytes) {
    char* p = ws;
    ws += (bytes + 255) & ~(size_t)255;
    return p;
  };
  float* pre_x  = (float*)carve((size_t)Nn * DIM * sizeof(float));
  float* self_x = (float*)carve((size_t)Nn * DIM * sizeof(float));
  float* acc    = (float*)carve((size_t)Nn * DIM * sizeof(float));
  _Float16* pWp0  = (_Float16*)carve(128 * 128 * 2);
  _Float16* pWp1  = (_Float16*)carve(128 * 128 * 2);
  _Float16* pngW1 = (_Float16*)carve(256 * 256 * 2);
  _Float16* pngW2 = (_Float16*)carve(256 * 256 * 2);
  _Float16* pWn0  = (_Float16*)carve(128 * 128 * 2);
  _Float16* pWn1  = (_Float16*)carve(128 * 128 * 2);
  _Float16* pWo0  = (_Float16*)carve(128 * 128 * 2);
  _Float16* pWo1  = (_Float16*)carve(128 * 128 * 2);
  _Float16* pFcn1 = (_Float16*)carve(32 * 32 * 2);
  _Float16* pFcn2 = (_Float16*)carve(32 * 512 * 2);
  _Float16* pL01  = (_Float16*)carve(384 * 32 * 2);
  _Float16* pL02  = (_Float16*)carve(32 * 512 * 2);

  auto pack = [&](const float* src, _Float16* dst, int K, int Nout, float sc) {
    int total = (K / 32) * (Nout / 16) * 32;
    pack_w_kernel<<<(total + 127) / 128, 128, 0, stream>>>(src, dst, K, Nout, sc);
  };
  const float s128 = 0.08838834764831845f;   // 1/sqrt(128)
  const float s32  = 0.17677669529663689f;   // 1/sqrt(32)
  const float s384 = 0.05103103630798288f;   // 1/sqrt(384)
  pack(Wp0, pWp0, 128, 128, s128);
  pack(Wp1, pWp1, 128, 128, s128);
  pack(ngW1, pngW1, 256, 256, 1.f);
  pack(ngW2, pngW2, 256, 256, 1.f);
  pack(Wn0, pWn0, 128, 128, s128);
  pack(Wn1, pWn1, 128, 128, s128);
  pack(Wo0, pWo0, 128, 128, s128);
  pack(Wo1, pWo1, 128, 128, s128);
  pack(fcnW1, pFcn1, 32, 32, s32);
  pack(fcnW2, pFcn2, 32, 512, s32);
  pack(l0W1, pL01, 384, 32, s384);
  pack(l0W2, pL02, 32, 512, s32);

  const int nFullN = Nn / 16, tailN = Nn % 16;
  const int nFullE = E / 16,  tailE = E % 16;

  if (nFullN)
    node_kernel<true><<<nFullN, 256, 0, stream>>>(
        x, pre_x, self_x, acc, pWp0, pWp1, pngW1, pngW2, pWn0, pWn1,
        bp0, ngb1, ngb2, bn0, 0, Nn);
  if (tailN)
    node_kernel<false><<<1, 256, 0, stream>>>(
        x, pre_x, self_x, acc, pWp0, pWp1, pngW1, pngW2, pWn0, pWn1,
        bp0, ngb1, ngb2, bn0, nFullN, Nn);

  if (nFullE)
    edge_kernel<true><<<nFullE, 256, 0, stream>>>(
        eidx, eattr, esh, pre_x, self_x, acc, pFcn1, pFcn2, pL01, pL02, 0, E);
  if (tailE)
    edge_kernel<false><<<1, 256, 0, stream>>>(
        eidx, eattr, esh, pre_x, self_x, acc, pFcn1, pFcn2, pL01, pL02,
        nFullE, E);

  if (nFullN)
    out_kernel<true><<<nFullN, 256, 0, stream>>>(
        acc, (float*)d_out, pWo0, pWo1, bo0, 0, Nn);
  if (tailN)
    out_kernel<false><<<1, 256, 0, stream>>>(
        acc, (float*)d_out, pWo0, pWo1, bo0, nFullN, Nn);

  (void)n_in; (void)out_size; (void)ws_size;
}
